// SwinTransformerBlock_30580167147545
// MI455X (gfx1250) — compile-verified, matrix operand order checked by reading, not run on hardware
//
#include <hip/hip_runtime.h>
#include <hip/hip_bf16.h>
#include <math.h>

// ---------- WMMA fragment helpers (gfx1250, wave32) ----------
typedef __attribute__((ext_vector_type(16))) _Float16 v16h;
typedef __attribute__((ext_vector_type(8)))  float    v8f;
typedef __attribute__((ext_vector_type(4)))  unsigned u32x4;
typedef __attribute__((ext_vector_type(4)))  int      v4i;

union FragU { v16h h; u32x4 q[2]; };

// 16-bit A/B fragment loader for V_WMMA_F32_16X16X32_F16.
// Matrix stored row-major [rows][ld] (for B this means N-major, i.e. W^T).
// Lane l holds row (base+(l&15)); chunk0 = K k0+(l>>4)*8, chunk1 = +16.
__device__ __forceinline__ v16h load_frag(const _Float16* p, int ld, int row, int k0, int lane) {
  const _Float16* base = p + (size_t)row * ld + k0 + ((lane >> 4) << 3);
  FragU f;
  f.q[0] = *(const u32x4*)(base);
  f.q[1] = *(const u32x4*)(base + 16);
  return f.h;
}

__device__ __forceinline__ v16h zfrag() {
  u32x4 z = {0u, 0u, 0u, 0u};
  FragU f; f.q[0] = z; f.q[1] = z; return f.h;
}

__device__ __forceinline__ v8f wmma_f16(v16h a, v16h b, v8f c) {
  return __builtin_amdgcn_wmma_f32_16x16x32_f16(false, a, false, b, (short)0, c, false, false);
}

// ---------- async-to-LDS path (gfx1250): guarded by __has_builtin ----------
#if defined(__has_builtin)
#if __has_builtin(__builtin_amdgcn_global_load_async_to_lds_b128)
#define HAVE_ASYNC_LDS 1
#endif
#endif
#ifndef HAVE_ASYNC_LDS
#define HAVE_ASYNC_LDS 0
#endif

#if HAVE_ASYNC_LDS
typedef __attribute__((address_space(1))) v4i as1_v4i;   // global int4*
typedef __attribute__((address_space(3))) v4i as3_v4i;   // LDS int4*
#endif

__device__ __forceinline__ void async_copy16(const _Float16* g, _Float16* l) {
#if HAVE_ASYNC_LDS
  __builtin_amdgcn_global_load_async_to_lds_b128(
      (as1_v4i*)(unsigned long long)(const void*)g,
      (as3_v4i*)(unsigned int)(unsigned long long)(void*)l, 0, 0);
#else
  *(u32x4*)l = *(const u32x4*)g;
#endif
}

__device__ __forceinline__ void async_wait_all() {
#if HAVE_ASYNC_LDS
  asm volatile("s_wait_asynccnt 0x0" ::: "memory");
#endif
}

// ---------- constants ----------
#define BB    32
#define CC    256
#define HH    56
#define HWSZ  3136          // 56*56
#define MTOK  100352        // 32*3136 (= 2048 windows * 49 tokens)
#define NQKV  768
#define NHID  1024

// ---------- GEMM core: block = 256 thr (8 waves), tile M=256 x N=64 ----------
// B tile (64 x 32 halves per K-step) double-buffered in LDS via async loads:
// 256 chunks of 16B == one b128 per thread per stage.
__device__ __forceinline__ void stageB(const _Float16* __restrict__ Bt, int K,
                                       int colBase, int k0,
                                       _Float16 (*bs)[40], int tid) {
  int n = tid >> 2;               // 0..63  (N row of the tile)
  int ch = (tid & 3) << 3;        // 0,8,16,24 halves
  const _Float16* g = Bt + (size_t)(colBase + n) * K + k0 + ch;
  async_copy16(g, &bs[n][ch]);
}

// acc[2][4]: two 16-row groups x four 16-col groups per wave (32x64 f32/wave).
__device__ __forceinline__ void gemm_core(const _Float16* __restrict__ A,
                                          const _Float16* __restrict__ Bt,
                                          int K, int mWaveBase, int colBase, int tid,
                                          _Float16 (*Bs)[64][40],
                                          v8f acc[2][4]) {
  int lane = tid & 31;
  int nk = K >> 5;
  int r0 = mWaveBase + (lane & 15);

  stageB(Bt, K, colBase, 0, Bs[0], tid);
  v16h a0 = load_frag(A, K, r0, 0, lane);
  v16h a1 = load_frag(A, K, r0 + 16, 0, lane);

  int buf = 0;
  for (int k = 0; k < nk; ++k) {
    async_wait_all();
    __syncthreads();                       // Bs[buf] ready for everyone
    if (k + 1 < nk) {
      stageB(Bt, K, colBase, (k + 1) << 5, Bs[buf ^ 1], tid);   // prefetch next B
      __builtin_prefetch(A + (size_t)r0 * K + ((k + 2) << 5), 0, 1);
    }
    v16h na0 = a0, na1 = a1;
    if (k + 1 < nk) {                      // pipeline next A fragments
      na0 = load_frag(A, K, r0, (k + 1) << 5, lane);
      na1 = load_frag(A, K, r0 + 16, (k + 1) << 5, lane);
    }
#pragma unroll
    for (int j = 0; j < 4; ++j) {
      v16h b = load_frag(&Bs[buf][0][0], 40, j * 16 + (lane & 15), 0, lane);
      acc[0][j] = wmma_f16(a0, b, acc[0][j]);
      acc[1][j] = wmma_f16(a1, b, acc[1][j]);
    }
    a0 = na0; a1 = na1;
    __syncthreads();                       // all waves done with Bs[buf]
    buf ^= 1;
  }
}

// ---------- 0) weight convert: wT[n][k] = (f16) w[k][n] ----------
__global__ void wconv_kernel(const float* __restrict__ w, _Float16* __restrict__ wt,
                             int K, int N) {
  size_t i = (size_t)blockIdx.x * 256 + threadIdx.x;
  if (i >= (size_t)K * N) return;
  int n = (int)(i / K), k = (int)(i % K);
  wt[i] = (_Float16)w[(size_t)k * N + n];
}

// ---------- 1) NCHW gather + LN1 + roll(-3,-3) + window partition -> y f16 ----------
__global__ void ln1_win_kernel(const float* __restrict__ x,
                               const float* __restrict__ g, const float* __restrict__ be,
                               _Float16* __restrict__ y) {
  int wave = threadIdx.x >> 5, lane = threadIdx.x & 31;
  int pos = blockIdx.x * 8 + wave;
  int b_ = pos / HWSZ; int hw = pos % HWSZ;
  int h = hw / HH, w = hw % HH;
  const float* xp = x + (size_t)b_ * CC * HWSZ + hw;
  float v[8]; float sum = 0.f;
#pragma unroll
  for (int i = 0; i < 8; ++i) { v[i] = xp[(size_t)(lane + 32 * i) * HWSZ]; sum += v[i]; }
#pragma unroll
  for (int off = 16; off > 0; off >>= 1) sum += __shfl_xor(sum, off, 32);
  float mean = sum * (1.f / 256.f);
  float var = 0.f;
#pragma unroll
  for (int i = 0; i < 8; ++i) { float d = v[i] - mean; var += d * d; }
#pragma unroll
  for (int off = 16; off > 0; off >>= 1) var += __shfl_xor(var, off, 32);
  float rstd = rsqrtf(var * (1.f / 256.f) + 1e-5f);
  int hd = h - 3; hd += (hd < 0) * HH;
  int wd = w - 3; wd += (wd < 0) * HH;
  int win = (b_ * 8 + hd / 7) * 8 + wd / 7;
  int tok = (hd % 7) * 7 + (wd % 7);
  _Float16* yr = y + ((size_t)win * 49 + tok) * CC;
#pragma unroll
  for (int i = 0; i < 8; ++i) {
    int c = lane + 32 * i;
    yr[c] = (_Float16)((v[i] - mean) * rstd * g[c] + be[c]);
  }
}

// ---------- 2) QKV GEMM: y[M,256] @ qkv_wT -> qkv f16 [M,768] ----------
__global__ void gemm_qkv_kernel(const _Float16* __restrict__ A, const _Float16* __restrict__ Bt,
                                const float* __restrict__ bias, _Float16* __restrict__ out) {
  __shared__ _Float16 Bs[2][64][40];
  int tid = threadIdx.x, wave = tid >> 5, lane = tid & 31;
  int mBase = blockIdx.y * 256 + wave * 32;
  int colBase = blockIdx.x * 64;
  v8f acc[2][4] = {};
  gemm_core(A, Bt, 256, mBase, colBase, tid, Bs, acc);
  int mOff = (lane >> 4) << 3;
#pragma unroll
  for (int j = 0; j < 4; ++j) {
    int col = colBase + j * 16 + (lane & 15);
    float bb = bias[col];
#pragma unroll
    for (int rg = 0; rg < 2; ++rg)
#pragma unroll
      for (int r = 0; r < 8; ++r) {
        int m = mBase + rg * 16 + r + mOff;
        out[(size_t)m * NQKV + col] = (_Float16)(acc[rg][j][r] + bb);
      }
  }
}

// ---------- 3) attention: one block (128 thr / 4 waves) per (window, head) ----------
__global__ void attn_kernel(const _Float16* __restrict__ qkv, _Float16* __restrict__ o) {
  int wh = blockIdx.x;
  int win = wh >> 3, head = wh & 7;
  const _Float16* base = qkv + (size_t)win * 49 * NQKV;
  const _Float16* qg = base + head * 32;
  const _Float16* kg = base + 256 + head * 32;
  const _Float16* vg = base + 512 + head * 32;

  __shared__ _Float16 ks[64][40];   // k, N-major [token][d], zero-padded tokens
  __shared__ _Float16 vt[32][72];   // v^T [d][token], zero-padded tokens
  __shared__ float    Sf[64][68];   // scores f32
  __shared__ _Float16 Pm[64][72];   // softmax probs f16

  int tid = threadIdx.x;
  for (int idx = tid; idx < 64 * 32; idx += 128) {
    int t = idx >> 5, d = idx & 31;
    _Float16 kv = (_Float16)0.f, vv = (_Float16)0.f;
    if (t < 49) { kv = kg[(size_t)t * NQKV + d]; vv = vg[(size_t)t * NQKV + d]; }
    ks[t][d] = kv;
    vt[d][t] = vv;
  }
  __syncthreads();

  int wave = tid >> 5, lane = tid & 31;
  int r0 = wave * 16;
  int qrow = r0 + (lane & 15);
  v16h a = (qrow < 49) ? load_frag(qg, NQKV, qrow, 0, lane) : zfrag();

  v8f s[4] = {};
#pragma unroll
  for (int j = 0; j < 4; ++j) {
    v16h b = load_frag(&ks[0][0], 40, j * 16 + (lane & 15), 0, lane);
    s[j] = wmma_f16(a, b, s[j]);
  }
  const float scale = 0.17677669529663687f;   // 32^-0.5
  int mOff = (lane >> 4) << 3;
#pragma unroll
  for (int j = 0; j < 4; ++j)
#pragma unroll
    for (int r = 0; r < 8; ++r)
      Sf[r0 + r + mOff][j * 16 + (lane & 15)] = s[j][r] * scale;
  __syncthreads();

  if (tid < 64) {
    int row = tid;
    if (row < 49) {
      float mx = -1e30f;
      for (int c = 0; c < 49; ++c) mx = fmaxf(mx, Sf[row][c]);
      float sum = 0.f;
      for (int c = 0; c < 49; ++c) { float e = expf(Sf[row][c] - mx); Sf[row][c] = e; sum += e; }
      float inv = 1.f / sum;
      for (int c = 0; c < 64; ++c)
        Pm[row][c] = (c < 49) ? (_Float16)(Sf[row][c] * inv) : (_Float16)0.f;
    } else {
      for (int c = 0; c < 64; ++c) Pm[row][c] = (_Float16)0.f;
    }
  }
  __syncthreads();

  v8f oa[2] = {};
#pragma unroll
  for (int k0 = 0; k0 < 64; k0 += 32) {
    v16h pa = load_frag(&Pm[0][0], 72, r0 + (lane & 15), k0, lane);
#pragma unroll
    for (int j = 0; j < 2; ++j) {
      v16h vb = load_frag(&vt[0][0], 72, j * 16 + (lane & 15), k0, lane);
      oa[j] = wmma_f16(pa, vb, oa[j]);
    }
  }
#pragma unroll
  for (int j = 0; j < 2; ++j) {
#pragma unroll
    for (int r = 0; r < 8; ++r) {
      int m = r0 + r + mOff;
      if (m < 49)
        o[((size_t)win * 49 + m) * CC + head * 32 + j * 16 + (lane & 15)] = (_Float16)oa[j][r];
    }
  }
}

// ---------- 4) proj GEMM + window reverse + roll(+3) + NCHW shortcut -> x1 f32 (NHWC) ----------
__global__ void gemm_proj_kernel(const _Float16* __restrict__ A, const _Float16* __restrict__ Bt,
                                 const float* __restrict__ bias, const float* __restrict__ x,
                                 float* __restrict__ x1) {
  __shared__ _Float16 Bs[2][64][40];
  int tid = threadIdx.x, wave = tid >> 5, lane = tid & 31;
  int mBase = blockIdx.y * 256 + wave * 32;
  int colBase = blockIdx.x * 64;
  v8f acc[2][4] = {};
  gemm_core(A, Bt, 256, mBase, colBase, tid, Bs, acc);
  int mOff = (lane >> 4) << 3;
#pragma unroll
  for (int j = 0; j < 4; ++j) {
    int col = colBase + j * 16 + (lane & 15);
    float bb = bias[col];
#pragma unroll
    for (int rg = 0; rg < 2; ++rg)
#pragma unroll
      for (int r = 0; r < 8; ++r) {
        int m = mBase + rg * 16 + r + mOff;     // window-order row
        int win = m / 49, tok = m % 49;
        int b_ = win >> 6, whi = (win >> 3) & 7, wwi = win & 7;
        int hd = whi * 7 + tok / 7, wd = wwi * 7 + tok % 7;
        int h = hd + 3; h -= (h >= HH) * HH;
        int w = wd + 3; w -= (w >= HH) * HH;
        size_t hw = (size_t)h * HH + w;
        float sc = x[((size_t)b_ * CC + col) * HWSZ + hw];
        x1[((size_t)b_ * HWSZ + hw) * CC + col] = acc[rg][j][r] + bb + sc;
      }
  }
}

// ---------- 5) LN2 over x1 rows -> h2 f16 ----------
__global__ void ln2_kernel(const float* __restrict__ x1,
                           const float* __restrict__ g, const float* __restrict__ be,
                           _Float16* __restrict__ h2) {
  int wave = threadIdx.x >> 5, lane = threadIdx.x & 31;
  int row = blockIdx.x * 8 + wave;
  const float* xp = x1 + (size_t)row * CC;
  float v[8]; float sum = 0.f;
#pragma unroll
  for (int i = 0; i < 8; ++i) { v[i] = xp[lane + 32 * i]; sum += v[i]; }
#pragma unroll
  for (int off = 16; off > 0; off >>= 1) sum += __shfl_xor(sum, off, 32);
  float mean = sum * (1.f / 256.f);
  float var = 0.f;
#pragma unroll
  for (int i = 0; i < 8; ++i) { float d = v[i] - mean; var += d * d; }
#pragma unroll
  for (int off = 16; off > 0; off >>= 1) var += __shfl_xor(var, off, 32);
  float rstd = rsqrtf(var * (1.f / 256.f) + 1e-5f);
  _Float16* hp = h2 + (size_t)row * CC;
#pragma unroll
  for (int i = 0; i < 8; ++i) {
    int c = lane + 32 * i;
    hp[c] = (_Float16)((v[i] - mean) * rstd * g[c] + be[c]);
  }
}

// ---------- 6) MLP1 GEMM + exact GELU -> hid f16 [M,1024] ----------
__global__ void gemm_mlp1_kernel(const _Float16* __restrict__ A, const _Float16* __restrict__ Bt,
                                 const float* __restrict__ bias, _Float16* __restrict__ out) {
  __shared__ _Float16 Bs[2][64][40];
  int tid = threadIdx.x, wave = tid >> 5, lane = tid & 31;
  int mBase = blockIdx.y * 256 + wave * 32;
  int colBase = blockIdx.x * 64;
  v8f acc[2][4] = {};
  gemm_core(A, Bt, 256, mBase, colBase, tid, Bs, acc);
  int mOff = (lane >> 4) << 3;
#pragma unroll
  for (int j = 0; j < 4; ++j) {
    int col = colBase + j * 16 + (lane & 15);
    float bb = bias[col];
#pragma unroll
    for (int rg = 0; rg < 2; ++rg)
#pragma unroll
      for (int r = 0; r < 8; ++r) {
        int m = mBase + rg * 16 + r + mOff;
        float t = acc[rg][j][r] + bb;
        float ge = 0.5f * t * (1.f + erff(t * 0.7071067811865475f));
        out[(size_t)m * NHID + col] = (_Float16)ge;
      }
  }
}

// ---------- 7) MLP2 GEMM + bias + x1 residual + NCHW scatter -> out f32 ----------
__global__ void gemm_mlp2_kernel(const _Float16* __restrict__ A, const _Float16* __restrict__ Bt,
                                 const float* __restrict__ bias, const float* __restrict__ x1,
                                 float* __restrict__ out) {
  __shared__ _Float16 Bs[2][64][40];
  int tid = threadIdx.x, wave = tid >> 5, lane = tid & 31;
  int mBase = blockIdx.y * 256 + wave * 32;
  int colBase = blockIdx.x * 64;
  v8f acc[2][4] = {};
  gemm_core(A, Bt, 1024, mBase, colBase, tid, Bs, acc);
  int mOff = (lane >> 4) << 3;
#pragma unroll
  for (int j = 0; j < 4; ++j) {
    int col = colBase + j * 16 + (lane & 15);
    float bb = bias[col];
#pragma unroll
    for (int rg = 0; rg < 2; ++rg)
#pragma unroll
      for (int r = 0; r < 8; ++r) {
        int m = mBase + rg * 16 + r + mOff;     // spatial-order row
        int b_ = m / HWSZ; int hw = m % HWSZ;
        float val = acc[rg][j][r] + bb + x1[(size_t)m * CC + col];
        out[((size_t)b_ * CC + col) * HWSZ + hw] = val;
      }
  }
}

// ---------- host ----------
extern "C" void kernel_launch(void* const* d_in, const int* in_sizes, int n_in,
                              void* d_out, int out_size, void* d_ws, size_t ws_size,
                              hipStream_t stream) {
  const float* x      = (const float*)d_in[0];
  const float* n1_g   = (const float*)d_in[1];
  const float* n1_b   = (const float*)d_in[2];
  const float* qkv_w  = (const float*)d_in[3];
  const float* qkv_b  = (const float*)d_in[4];
  const float* proj_w = (const float*)d_in[5];
  const float* proj_b = (const float*)d_in[6];
  const float* n2_g   = (const float*)d_in[7];
  const float* n2_b   = (const float*)d_in[8];
  const float* mlp_w1 = (const float*)d_in[9];
  const float* mlp_b1 = (const float*)d_in[10];
  const float* mlp_w2 = (const float*)d_in[11];
  const float* mlp_b2 = (const float*)d_in[12];
  float* out = (float*)d_out;

  char* p = (char*)d_ws;
  size_t off = 0;
  auto take = [&](size_t bytes) -> void* {
    void* r = p + off;
    off = (off + bytes + 255) & ~(size_t)255;
    return r;
  };
  _Float16* qkv_wT = (_Float16*)take((size_t)NQKV * CC * 2);
  _Float16* projWT = (_Float16*)take((size_t)CC * CC * 2);
  _Float16* w1T    = (_Float16*)take((size_t)NHID * CC * 2);
  _Float16* w2T    = (_Float16*)take((size_t)CC * NHID * 2);
  _Float16* ybuf   = (_Float16*)take((size_t)MTOK * CC * 2);  // reused: y -> o -> h2
  _Float16* qkvB   = (_Float16*)take((size_t)MTOK * NQKV * 2);
  float*    x1     = (float*)take((size_t)MTOK * CC * 4);
  _Float16* hid    = (_Float16*)take((size_t)MTOK * NHID * 2);

  wconv_kernel<<<(CC * NQKV) / 256, 256, 0, stream>>>(qkv_w, qkv_wT, CC, NQKV);
  wconv_kernel<<<(CC * CC) / 256, 256, 0, stream>>>(proj_w, projWT, CC, CC);
  wconv_kernel<<<(CC * NHID) / 256, 256, 0, stream>>>(mlp_w1, w1T, CC, NHID);
  wconv_kernel<<<(NHID * CC) / 256, 256, 0, stream>>>(mlp_w2, w2T, NHID, CC);

  ln1_win_kernel<<<MTOK / 8, 256, 0, stream>>>(x, n1_g, n1_b, ybuf);

  gemm_qkv_kernel<<<dim3(NQKV / 64, MTOK / 256), 256, 0, stream>>>(ybuf, qkv_wT, qkv_b, qkvB);

  attn_kernel<<<2048 * 8, 128, 0, stream>>>(qkvB, ybuf);

  gemm_proj_kernel<<<dim3(CC / 64, MTOK / 256), 256, 0, stream>>>(ybuf, projWT, proj_b, x, x1);

  ln2_kernel<<<MTOK / 8, 256, 0, stream>>>(x1, n2_g, n2_b, ybuf);

  gemm_mlp1_kernel<<<dim3(NHID / 64, MTOK / 256), 256, 0, stream>>>(ybuf, w1T, mlp_b1, hid);

  gemm_mlp2_kernel<<<dim3(CC / 64, MTOK / 256), 256, 0, stream>>>(hid, w2T, mlp_b2, x1, out);

  (void)in_sizes; (void)n_in; (void)out_size; (void)ws_size;
}
